// FastTypedLinear_80762565034485
// MI455X (gfx1250) — matrix operand order repaired and checked by reading, not compiled, not run on hardware
//
#include <hip/hip_runtime.h>

// Grouped GEMM: out[n] = x[n] @ W[types[n]] + b[types[n]]
// N=65536, IN=1024, OUT=1024, 8 types, types sorted.
//
// Engine: v_wmma_f32_16x16x32_bf16 with bf16 hi/lo split precision
// (hi*hi + hi*lo + lo*hi, f32 accumulate) -> near-f32 accuracy at ~3/8 the
// cost of the native f32 WMMA (16x16x4) path. Compute-bound: 137 GFLOP vs a
// ~24 us HBM floor (550 MB @ 23.3 TB/s).
//
// Fast path: W is pre-split into bf16 hi/lo ONCE per launch, stored in d_ws
// in this GEMM's contiguous-k B-fragment layout (per (type, ntile, kchunk): 32 lanes x
// 16 bf16 contiguous = 1 KB). 32 MB total -> L2-resident, so the GEMM loads
// B fragments straight from global (global_load_b128, fully coalesced) with
// zero per-block conversion VALU and no LDS round-trip for B.

#define IN_F 1024
#define OUT_F 1024
#define BLOCK_M 32
#define BLOCK_N 256
#define KC 32
#define NTHREADS 256
#define NT_N (OUT_F / 16)   // 64 column tiles
#define NT_K (IN_F / KC)    // 32 K chunks

typedef __bf16 v16bf __attribute__((ext_vector_type(16)));
typedef float  v8f   __attribute__((ext_vector_type(8)));
typedef unsigned int u32x4 __attribute__((ext_vector_type(4)));
typedef float        f32x4 __attribute__((ext_vector_type(4)));

union Frag { v16bf v; u32x4 q[2]; };

__device__ __forceinline__ unsigned short f2bf_rne(float f) {
    unsigned int u = __float_as_uint(f);
    unsigned int r = 0x7FFFu + ((u >> 16) & 1u);
    return (unsigned short)((u + r) >> 16);
}
__device__ __forceinline__ float bf2f(unsigned short h) {
    return __uint_as_float(((unsigned int)h) << 16);
}

// ---------------------------------------------------------------------------
// Pass 1: split W (f32 [T, K, N]) into bf16 hi/lo, stored pre-tiled in the
// B-fragment layout this GEMM loads (contiguous k, not AMD's table order):
//   linear = (((t*NT_N + ntile)*NT_K + kc)*32 + lane)*16
//   lane<16:  N = ntile*16 + lane,    K = kc*32 + e      (e = 0..15)
//   lane>=16: N = ntile*16 + lane-16, K = kc*32 + 16 + e
//   bf16 pair packing: element 2v -> bits[15:0], 2v+1 -> bits[31:16].
// ---------------------------------------------------------------------------
__global__ __launch_bounds__(NTHREADS)
void convert_w_tiled(const float* __restrict__ W,
                     unsigned short* __restrict__ whi,
                     unsigned short* __restrict__ wlo,
                     unsigned int total_threads)
{
    unsigned int gid = blockIdx.x * NTHREADS + threadIdx.x;
    if (gid >= total_threads) return;
    const int lane = gid & 31;
    unsigned int unit = gid >> 5;            // (t*NT_N + nt)*NT_K + kc
    const int kc = unit % NT_K;
    unsigned int tmp = unit / NT_K;
    const int nt = tmp % NT_N;
    const int t  = tmp / NT_N;

    const int n     = nt * 16 + (lane & 15);
    const int kbase = kc * KC + (lane >> 4) * 16;
    const float* src = W + (size_t)t * (IN_F * OUT_F) + (size_t)kbase * OUT_F + n;

    unsigned int hw[8], lw[8];
    #pragma unroll
    for (int i = 0; i < 8; ++i) {
        float a0 = src[(size_t)(2 * i) * OUT_F];       // coalesced across lanes (n)
        float a1 = src[(size_t)(2 * i + 1) * OUT_F];
        unsigned short h0 = f2bf_rne(a0), h1 = f2bf_rne(a1);
        unsigned short l0 = f2bf_rne(a0 - bf2f(h0));
        unsigned short l1 = f2bf_rne(a1 - bf2f(h1));
        hw[i] = (unsigned int)h0 | ((unsigned int)h1 << 16);
        lw[i] = (unsigned int)l0 | ((unsigned int)l1 << 16);
    }
    const size_t dst = (size_t)gid * 16;     // elements
    u32x4 a = {hw[0], hw[1], hw[2], hw[3]};
    u32x4 b = {hw[4], hw[5], hw[6], hw[7]};
    u32x4 c = {lw[0], lw[1], lw[2], lw[3]};
    u32x4 d = {lw[4], lw[5], lw[6], lw[7]};
    *(u32x4*)(whi + dst)     = a;
    *(u32x4*)(whi + dst + 8) = b;
    *(u32x4*)(wlo + dst)     = c;
    *(u32x4*)(wlo + dst + 8) = d;
}

// ---------------------------------------------------------------------------
// Pass 2: GEMM. Block = 32 rows x 256 cols, 8 waves (2 x 4); each wave owns
// 16 rows x 64 cols = 4 C tiles -> 12 WMMAs per 32-K chunk. A goes through
// LDS (masked + hi/lo split, 1 float4/thread/chunk); B fragments come
// straight from the tiled workspace via global_load_b128 (L2-resident).
// ---------------------------------------------------------------------------
__global__ __launch_bounds__(NTHREADS)
void typed_linear_wmma_fast(const float* __restrict__ x,
                            const int*   __restrict__ types,
                            const unsigned short* __restrict__ whi,
                            const unsigned short* __restrict__ wlo,
                            const float* __restrict__ bias,
                            float* __restrict__ out)
{
    __shared__ __align__(16) unsigned short sAhi[BLOCK_M][KC];
    __shared__ __align__(16) unsigned short sAlo[BLOCK_M][KC];
    __shared__ int sTypes[BLOCK_M];

    const int tid  = threadIdx.x;
    const int lane = tid & 31;
    const int wave = tid >> 5;
    const int mi   = wave >> 2;          // 0..1 -> 16-row slab
    const int ni   = wave & 3;           // 0..3 -> 64-col slab
    const int lh   = lane & 15;
    const int lu   = lane >> 4;

    const int row0 = blockIdx.x * BLOCK_M;
    const int col0 = blockIdx.y * BLOCK_N;

    if (tid < BLOCK_M) sTypes[tid] = types[row0 + tid];
    __syncthreads();
    const int tmin = sTypes[0];
    const int tmax = sTypes[BLOCK_M - 1];   // sorted -> types in tile span [tmin,tmax]

    v8f acc[4];
    #pragma unroll
    for (int c = 0; c < 4; ++c) acc[c] = (v8f){0.f,0.f,0.f,0.f,0.f,0.f,0.f,0.f};

    // A fragment (16x32 bf16): lanes 0-15 K 0-15; lanes 16-31 K 16-31
    const int a_row = mi * 16 + lh;
    const int a_k0  = lu * 16;

    // A staging: 256 float4 units, one per thread
    const int sa_r = tid >> 3;
    const int sa_k = (tid & 7) * 4;

    const int ntw = (col0 >> 4) + ni * 4;   // first of this wave's 4 column tiles

    for (int t = tmin; t <= tmax; ++t) {
        for (int kc = 0; kc < NT_K; ++kc) {
            const int k0 = kc * KC;
            __syncthreads();
            // ---- stage A chunk (mask rows whose type != t) ----
            {
                f32x4 v = *(const f32x4*)(x + (size_t)(row0 + sa_r) * IN_F + (k0 + sa_k));
                const bool act = (sTypes[sa_r] == t);
                #pragma unroll
                for (int i = 0; i < 4; ++i) {
                    float a = act ? v[i] : 0.0f;
                    unsigned short hi = f2bf_rne(a);
                    unsigned short lo = f2bf_rne(a - bf2f(hi));
                    sAhi[sa_r][sa_k + i] = hi;
                    sAlo[sa_r][sa_k + i] = lo;
                }
            }
            if (k0 + KC < IN_F)
                __builtin_prefetch(x + (size_t)(row0 + sa_r) * IN_F + (k0 + KC + sa_k), 0, 1);
            __syncthreads();

            // ---- A fragments from LDS ----
            Frag ahi, alo;
            ahi.q[0] = *(const u32x4*)&sAhi[a_row][a_k0];
            ahi.q[1] = *(const u32x4*)&sAhi[a_row][a_k0 + 8];
            alo.q[0] = *(const u32x4*)&sAlo[a_row][a_k0];
            alo.q[1] = *(const u32x4*)&sAlo[a_row][a_k0 + 8];

            // ---- B fragments straight from the tiled workspace ----
            Frag bh[4], bl[4];
            #pragma unroll
            for (int c = 0; c < 4; ++c) {
                const size_t off =
                    ((((size_t)t * NT_N + (ntw + c)) * NT_K + kc) * 32 + lane) * 16;
                bh[c].q[0] = *(const u32x4*)(whi + off);
                bh[c].q[1] = *(const u32x4*)(whi + off + 8);
                bl[c].q[0] = *(const u32x4*)(wlo + off);
                bl[c].q[1] = *(const u32x4*)(wlo + off + 8);
                if (kc + 1 < NT_K) {
                    __builtin_prefetch(whi + off + 512, 0, 1);   // next K chunk
                    __builtin_prefetch(wlo + off + 512, 0, 1);
                }
            }

            // ---- bf16x3: hi*hi + hi*lo + lo*hi ----
            #pragma unroll
            for (int c = 0; c < 4; ++c)
                acc[c] = __builtin_amdgcn_wmma_f32_16x16x32_bf16(false, ahi.v, false, bh[c].v, (short)0, acc[c], false, false);
            #pragma unroll
            for (int c = 0; c < 4; ++c)
                acc[c] = __builtin_amdgcn_wmma_f32_16x16x32_bf16(false, ahi.v, false, bl[c].v, (short)0, acc[c], false, false);
            #pragma unroll
            for (int c = 0; c < 4; ++c)
                acc[c] = __builtin_amdgcn_wmma_f32_16x16x32_bf16(false, alo.v, false, bh[c].v, (short)0, acc[c], false, false);
        }
    }

    // ---- epilogue: bias from each row's own type, store ----
    // C layout: VGPR j -> M=j (lanes 0-15) / M=j+8 (lanes 16-31), N=lane&15.
    const int c_col = col0 + ni * 64 + lh;
    #pragma unroll
    for (int j = 0; j < 8; ++j) {
        const int rl = mi * 16 + lu * 8 + j;
        const int r  = row0 + rl;
        const int ty = sTypes[rl];
        const float* bt  = bias + (size_t)ty * OUT_F;
        float* orow = out + (size_t)r * OUT_F;
        #pragma unroll
        for (int c = 0; c < 4; ++c)
            orow[c_col + c * 16] = acc[c][j] + bt[c_col + c * 16];
    }
}

// ---------------------------------------------------------------------------
// Fallback (no workspace): self-contained kernel, inline W conversion via LDS.
// ---------------------------------------------------------------------------
#define FB_BLOCK_N 128
__global__ __launch_bounds__(NTHREADS)
void typed_linear_wmma_fb(const float* __restrict__ x,
                          const int*   __restrict__ types,
                          const float* __restrict__ W,
                          const float* __restrict__ bias,
                          float* __restrict__ out)
{
    __shared__ __align__(16) unsigned short sAhi[BLOCK_M][KC];
    __shared__ __align__(16) unsigned short sAlo[BLOCK_M][KC];
    __shared__ __align__(16) unsigned short sBhi[FB_BLOCK_N][KC];
    __shared__ __align__(16) unsigned short sBlo[FB_BLOCK_N][KC];
    __shared__ int sTypes[BLOCK_M];

    const int tid  = threadIdx.x;
    const int lane = tid & 31;
    const int wave = tid >> 5;
    const int mi   = wave >> 2;
    const int ni   = wave & 3;
    const int lh   = lane & 15;
    const int lu   = lane >> 4;

    const int row0 = blockIdx.x * BLOCK_M;
    const int col0 = blockIdx.y * FB_BLOCK_N;

    if (tid < BLOCK_M) sTypes[tid] = types[row0 + tid];
    __syncthreads();
    const int tmin = sTypes[0];
    const int tmax = sTypes[BLOCK_M - 1];

    v8f acc0 = {0.f,0.f,0.f,0.f,0.f,0.f,0.f,0.f};
    v8f acc1 = acc0;

    const int a_row = mi * 16 + lh;
    const int a_k0  = lu * 16;
    const int b_n0  = ni * 32 + lh;
    const int b_k0  = lu * 16;
    const int sa_r = tid >> 3;
    const int sa_k = (tid & 7) * 4;

    for (int t = tmin; t <= tmax; ++t) {
        const float* __restrict__ Wt = W + (size_t)t * (IN_F * OUT_F);
        for (int k0 = 0; k0 < IN_F; k0 += KC) {
            __syncthreads();
            {
                f32x4 v = *(const f32x4*)(x + (size_t)(row0 + sa_r) * IN_F + (k0 + sa_k));
                const bool act = (sTypes[sa_r] == t);
                #pragma unroll
                for (int i = 0; i < 4; ++i) {
                    float a = act ? v[i] : 0.0f;
                    unsigned short hi = f2bf_rne(a);
                    unsigned short lo = f2bf_rne(a - bf2f(hi));
                    sAhi[sa_r][sa_k + i] = hi;
                    sAlo[sa_r][sa_k + i] = lo;
                }
            }
            #pragma unroll
            for (int u = 0; u < 4; ++u) {
                int unit = tid + u * NTHREADS;
                int kk = unit >> 5;
                int nn = (unit & 31) * 4;
                f32x4 v = *(const f32x4*)(Wt + (size_t)(k0 + kk) * OUT_F + (col0 + nn));
                #pragma unroll
                for (int i = 0; i < 4; ++i) {
                    float a = v[i];
                    unsigned short hi = f2bf_rne(a);
                    unsigned short lo = f2bf_rne(a - bf2f(hi));
                    sBhi[nn + i][kk] = hi;
                    sBlo[nn + i][kk] = lo;
                }
            }
            __syncthreads();

            Frag ahi, alo;
            ahi.q[0] = *(const u32x4*)&sAhi[a_row][a_k0];
            ahi.q[1] = *(const u32x4*)&sAhi[a_row][a_k0 + 8];
            alo.q[0] = *(const u32x4*)&sAlo[a_row][a_k0];
            alo.q[1] = *(const u32x4*)&sAlo[a_row][a_k0 + 8];

            Frag bh0, bl0, bh1, bl1;
            bh0.q[0] = *(const u32x4*)&sBhi[b_n0][b_k0];
            bh0.q[1] = *(const u32x4*)&sBhi[b_n0][b_k0 + 8];
            bl0.q[0] = *(const u32x4*)&sBlo[b_n0][b_k0];
            bl0.q[1] = *(const u32x4*)&sBlo[b_n0][b_k0 + 8];
            bh1.q[0] = *(const u32x4*)&sBhi[b_n0 + 16][b_k0];
            bh1.q[1] = *(const u32x4*)&sBhi[b_n0 + 16][b_k0 + 8];
            bl1.q[0] = *(const u32x4*)&sBlo[b_n0 + 16][b_k0];
            bl1.q[1] = *(const u32x4*)&sBlo[b_n0 + 16][b_k0 + 8];

            acc0 = __builtin_amdgcn_wmma_f32_16x16x32_bf16(false, ahi.v, false, bh0.v, (short)0, acc0, false, false);
            acc1 = __builtin_amdgcn_wmma_f32_16x16x32_bf16(false, ahi.v, false, bh1.v, (short)0, acc1, false, false);
            acc0 = __builtin_amdgcn_wmma_f32_16x16x32_bf16(false, ahi.v, false, bl0.v, (short)0, acc0, false, false);
            acc1 = __builtin_amdgcn_wmma_f32_16x16x32_bf16(false, ahi.v, false, bl1.v, (short)0, acc1, false, false);
            acc0 = __builtin_amdgcn_wmma_f32_16x16x32_bf16(false, alo.v, false, bh0.v, (short)0, acc0, false, false);
            acc1 = __builtin_amdgcn_wmma_f32_16x16x32_bf16(false, alo.v, false, bh1.v, (short)0, acc1, false, false);
        }
    }

    const int c_col = col0 + ni * 32 + lh;
    #pragma unroll
    for (int j = 0; j < 8; ++j) {
        const int rl = mi * 16 + lu * 8 + j;
        const int r  = row0 + rl;
        const int ty = sTypes[rl];
        const float* bt = bias + (size_t)ty * OUT_F;
        out[(size_t)r * OUT_F + c_col]      = acc0[j] + bt[c_col];
        out[(size_t)r * OUT_F + c_col + 16] = acc1[j] + bt[c_col + 16];
    }
}

extern "C" void kernel_launch(void* const* d_in, const int* in_sizes, int n_in,
                              void* d_out, int out_size, void* d_ws, size_t ws_size,
                              hipStream_t stream) {
    const float* x     = (const float*)d_in[0];
    const int*   types = (const int*)d_in[1];
    const float* W     = (const float*)d_in[2];
    const float* b     = (const float*)d_in[3];
    float* out = (float*)d_out;
    const int n      = in_sizes[1];                       // rows
    const int ntypes = in_sizes[2] / (IN_F * OUT_F);      // 8

    const size_t welems = (size_t)ntypes * IN_F * OUT_F;  // per hi or lo
    if (ws_size >= 2 * welems * sizeof(unsigned short)) {
        unsigned short* whi = (unsigned short*)d_ws;
        unsigned short* wlo = whi + welems;
        const unsigned int total = (unsigned int)ntypes * NT_N * NT_K * 32;
        convert_w_tiled<<<dim3((total + NTHREADS - 1) / NTHREADS), dim3(NTHREADS), 0, stream>>>(
            W, whi, wlo, total);
        dim3 grid(n / BLOCK_M, OUT_F / BLOCK_N);
        typed_linear_wmma_fast<<<grid, dim3(NTHREADS), 0, stream>>>(x, types, whi, wlo, b, out);
    } else {
        dim3 grid(n / BLOCK_M, OUT_F / FB_BLOCK_N);
        typed_linear_wmma_fb<<<grid, dim3(NTHREADS), 0, stream>>>(x, types, W, b, out);
    }
    (void)n_in; (void)out_size;
}
